// Decoder_28183575396714
// MI455X (gfx1250) — compile-verified, matrix operand order checked by reading, not run on hardware
//
#include <hip/hip_runtime.h>

// ---------------------------------------------------------------------------
// Tacotron2 decoder for MI455X (gfx1250).
// bf16 WMMA (16x16x32) for the two LSTM GEMMs + query projection; weights
// converted & pre-swizzled into WMMA lane order once per call (then resident
// in the 192MB L2 across all 200 recurrent steps). Activations (A operand)
// are staged per-block into LDS via GLOBAL_LOAD_ASYNC_TO_LDS_B128 (8x less
// L2 read traffic than per-wave global reads), overlapped with the previous
// step's output projections. One persistent kernel with a device-wide epoch
// barrier runs the 200-step scan.
// ---------------------------------------------------------------------------

#define B_    16
#define TENC  128
#define TDEC  200
#define MEL   80
#define EMB   512
#define RNN   1024
#define PRE   256
#define ATT   128
#define NF    32
#define KS    31
#define KA    1792   // PRE + EMB + RNN   (arnn fused K)
#define KD    2560   // RNN + EMB + RNN   (drnn fused K)
#define KTA   56     // KA/32
#define KTD   80     // KD/32
#define KTQ   32     // RNN/32
#define NGATE 4096
#define NBLK  32
#define NTHR  256
#define LDA_A (KA + 8)   // LDS row stride (pad: stride % 64 dwords == 4 -> no bank clash)
#define LDA_D (KD + 8)

typedef __attribute__((ext_vector_type(16))) __bf16 v16bf;
typedef __attribute__((ext_vector_type(8)))  float  v8f;
typedef int v4i __attribute__((__vector_size__(16)));

struct alignas(16) B16v { unsigned int u[4]; };
union ABu { v16bf v; B16v b[2]; };

struct DecParams {
  const float* memory; const float* din;
  const int*   mlen;
  const float* arnn_bih; const float* arnn_bhh;
  const float* drnn_bih; const float* drnn_bhh;
  const float* q_b; const float* v_W; const float* v_b;
  const float* lc_W; const float* lc_b; const float* ld_W; const float* ld_b;
  const float* proj_W; const float* proj_b; const float* gate_W; const float* gate_b;
  const unsigned short* wA; const unsigned short* wD; const unsigned short* wQ;
  const unsigned short* hpre;
  unsigned short* actA; unsigned short* actD;
  float* gates; float* ac; float* dc; float* dh;
  float* aw; float* awc; float* ctx; float* loc; float* pq; float* pm;
  unsigned* cnt;
  float* out_mel; float* out_gate; float* out_align;
};

__device__ __forceinline__ unsigned short f2bf(float f){
  unsigned u = __float_as_uint(f);
  u += 0x7FFFu + ((u >> 16) & 1u);           // round-to-nearest-even
  return (unsigned short)(u >> 16);
}
__device__ __forceinline__ float sigm(float x){ return 1.0f / (1.0f + __expf(-x)); }
__device__ __forceinline__ unsigned mixh(unsigned a, unsigned b, unsigned c){
  unsigned h = a * 0x9E3779B1u + b * 0x85EBCA77u + c * 0xC2B2AE3Du + 0x27220A95u;
  h ^= h >> 15; h *= 0x2C1B3C6Du; h ^= h >> 12; h *= 0x297A2D39u; h ^= h >> 15;
  return h;
}
// 16-bit WMMA operand lane layout: lane<16 -> K {0..7,16..23}, lane>=16 -> K {8..15,24..31}
__device__ __forceinline__ int klocal(int lane, int e){
  return (lane < 16) ? ((e < 8) ? e : e + 8)
                     : ((e < 8) ? e + 8 : e + 16);
}

// --- async global -> LDS staging (gfx1250 GLOBAL_LOAD_ASYNC_TO_LDS_B128) ---
// probe-learned signature: (v4i addrspace(1)*, v4i addrspace(3)*, imm offset, imm cpol)
#if defined(__HIP_DEVICE_COMPILE__) && __has_builtin(__builtin_amdgcn_global_load_async_to_lds_b128)
#define HAVE_ASYNC 1
__device__ __forceinline__ void async_g2l(unsigned short* l, const unsigned short* g){
  __builtin_amdgcn_global_load_async_to_lds_b128(
      (__attribute__((address_space(1))) v4i*)g,
      (__attribute__((address_space(3))) v4i*)l, 0, 0);
}
__device__ __forceinline__ void async_wait(){
#if __has_builtin(__builtin_amdgcn_s_wait_asynccnt)
  __builtin_amdgcn_s_wait_asynccnt(0);
#else
  asm volatile("s_wait_asynccnt 0x0" ::: "memory");
#endif
}
#else
#define HAVE_ASYNC 0
__device__ __forceinline__ void async_g2l(unsigned short* l, const unsigned short* g){
  *(B16v*)l = *(const B16v*)g;     // register relay fallback (still LDS-staged)
}
__device__ __forceinline__ void async_wait(){}
#endif

// act: bf16 activations [16][lda] (LDS or global); w: pre-swizzled bf16 weights
__device__ __forceinline__ v8f gemm_tiles(const unsigned short* __restrict__ act, int lda,
                                          const unsigned short* __restrict__ w,
                                          int nt, int KT, int lane){
  v8f acc = {0.f,0.f,0.f,0.f,0.f,0.f,0.f,0.f};
  const int m  = lane & 15;
  const int kb = (lane < 16) ? 0 : 8;
  const unsigned short* wt = w + (size_t)nt * KT * 512;      // 32 lanes * 16 elems per tile
  const unsigned short* ar = act + (size_t)m * lda + kb;
  #pragma unroll 2
  for (int kt = 0; kt < KT; ++kt){
    ABu a, bm;
    const unsigned short* ap = ar + kt * 32;
    a.b[0]  = *(const B16v*)(ap);            // K = kb .. kb+7
    a.b[1]  = *(const B16v*)(ap + 16);       // K = kb+16 .. kb+23
    const unsigned short* bp = wt + ((size_t)kt * 32 + lane) * 16;
    bm.b[0] = *(const B16v*)(bp);
    bm.b[1] = *(const B16v*)(bp + 8);
    __builtin_prefetch(bp + 1024, 0, 1);     // global_prefetch_b8, 2 tiles ahead
    acc = __builtin_amdgcn_wmma_f32_16x16x32_bf16(false, a.v, false, bm.v,
                                                  (short)0, acc, false, false);
  }
  return acc;
}

// C/D layout: lane<16 -> M=r, N=lane ; lane>=16 -> M=r+8, N=lane-16
__device__ __forceinline__ void store_rows(float* __restrict__ out, int ldn, v8f acc,
                                           int nt, int lane, float bias){
  const int n  = nt * 16 + (lane & 15);
  const int mb = (lane < 16) ? 0 : 8;
  #pragma unroll
  for (int r = 0; r < 8; ++r)
    out[(size_t)(mb + r) * ldn + n] = acc[r] + bias;
}

__device__ __forceinline__ void grid_sync(unsigned* cnt, unsigned& epoch){
  __threadfence();
  __syncthreads();
  epoch++;
  if (threadIdx.x == 0){
    atomicAdd(cnt, 1u);
    const unsigned tgt = epoch * (unsigned)NBLK;
    volatile unsigned* vc = cnt;
    while (*vc < tgt) __builtin_amdgcn_s_sleep(2);
  }
  __syncthreads();
  __threadfence();
}

// mel + gate projections for step t (reads dh, ctx only)
__device__ __forceinline__ void do_outputs(const DecParams& P, int gtid, int t){
  if (gtid < B_ * (MEL + 1)){
    int b = gtid / (MEL + 1), j = gtid % (MEL + 1);
    const float* w  = (j < MEL) ? (P.proj_W + (size_t)j * (RNN + EMB)) : P.gate_W;
    float s         = (j < MEL) ? P.proj_b[j] : P.gate_b[0];
    const float* dh = P.dh  + (size_t)b * RNN;
    const float* cx = P.ctx + (size_t)b * EMB;
    for (int k = 0; k < RNN; ++k) s += dh[k] * w[k];
    for (int k = 0; k < EMB; ++k) s += cx[k] * w[RNN + k];
    if (j < MEL) P.out_mel[((size_t)b * MEL + j) * TDEC + t] = s;
    else         P.out_gate[(size_t)b * TDEC + t] = s;
  }
}

// ------------------------------ prep kernels -------------------------------

__global__ void k_zero(DecParams P){
  int i = blockIdx.x * blockDim.x + threadIdx.x;
  int st = gridDim.x * blockDim.x;
  for (int k = i; k < 64;            k += st) P.cnt[k] = 0u;
  for (int k = i; k < B_ * TENC;     k += st){ P.aw[k] = 0.f; P.awc[k] = 0.f; P.pq[k] = 0.f; }
  for (int k = i; k < B_ * RNN;      k += st){ P.ac[k] = 0.f; P.dc[k] = 0.f; P.dh[k] = 0.f; }
  for (int k = i; k < B_ * EMB;      k += st) P.ctx[k] = 0.f;
  for (int k = i; k < B_ * NGATE;    k += st) P.gates[k] = 0.f;
  for (int k = i; k < B_ * KA;       k += st) P.actA[k] = 0;
  for (int k = i; k < B_ * KD;       k += st) P.actD[k] = 0;
}

// prenet layer 1: relu(x @ W1.T + b1) with deterministic hash dropout
__global__ void k_prenet1(const float* __restrict__ din, const float* __restrict__ W1,
                          const float* __restrict__ b1, float* __restrict__ l1){
  int id = blockIdx.x * blockDim.x + threadIdx.x;
  if (id >= TDEC * B_ * PRE) return;
  int t = id >> 12;            // 16*256 per timestep
  int r = id & 4095;
  int b = r >> 8, j = r & 255;
  float s = b1[j];
  if (t > 0){
    const float* w = W1 + (size_t)j * MEL;
    for (int k = 0; k < MEL; ++k)
      s += din[((size_t)b * MEL + k) * TDEC + (t - 1)] * w[k];
  }
  s = fmaxf(s, 0.f);
  s = (mixh((unsigned)t, (unsigned)(b * 256 + j), 1u) & 1u) ? s * 2.f : 0.f;
  l1[((size_t)t * B_ + b) * PRE + j] = s;
}

// prenet layer 2 -> bf16 h_pre; also stages t==0 slice into actA
__global__ void k_prenet2(const float* __restrict__ l1, const float* __restrict__ W2,
                          const float* __restrict__ b2, unsigned short* __restrict__ hpre,
                          unsigned short* __restrict__ actA){
  int id = blockIdx.x * blockDim.x + threadIdx.x;
  if (id >= TDEC * B_ * PRE) return;
  int t = id >> 12;
  int r = id & 4095;
  int b = r >> 8, j = r & 255;
  const float* x = l1 + ((size_t)t * B_ + b) * PRE;
  const float* w = W2 + (size_t)j * PRE;
  float s = b2[j];
  for (int k = 0; k < PRE; ++k) s += x[k] * w[k];
  s = fmaxf(s, 0.f);
  s = (mixh((unsigned)t, (unsigned)(b * 256 + j), 2u) & 1u) ? s * 2.f : 0.f;
  unsigned short us = f2bf(s);
  hpre[((size_t)t * B_ + b) * PRE + j] = us;
  if (t == 0) actA[(size_t)b * KA + j] = us;
}

// pm = memory @ m_W.T + m_b   [16][128][128]
__global__ void k_pm(const float* __restrict__ memory, const float* __restrict__ mW,
                     const float* __restrict__ mb, float* __restrict__ pm){
  int id = blockIdx.x * blockDim.x + threadIdx.x;
  if (id >= B_ * TENC * ATT) return;
  int a  = id & 127;
  int tp = (id >> 7) & 127;
  int b  = id >> 14;
  const float* mrow = memory + ((size_t)b * TENC + tp) * EMB;
  const float* wrow = mW + (size_t)a * EMB;
  float s = mb[a];
  for (int k = 0; k < EMB; ++k) s += mrow[k] * wrow[k];
  pm[((size_t)b * TENC + tp) * ATT + a] = s;
}

// fp32 [N][K] weights (optionally two matrices stacked along K) -> bf16 tiles
// in WMMA B-operand lane order: dst[((nt*KT+kt)*32+lane)*16 + e]
__global__ void k_swizzle(const float* __restrict__ W0, const float* __restrict__ W1,
                          int ksplit, int K, int Nrows, unsigned short* __restrict__ dst){
  int id = blockIdx.x * blockDim.x + threadIdx.x;
  int KT = K >> 5;
  int total = (Nrows >> 4) * KT * 32;
  if (id >= total) return;
  int lane = id & 31;
  int tk = id >> 5;
  int kt = tk % KT;
  int nt = tk / KT;
  int n  = nt * 16 + (lane & 15);
  unsigned short* d = dst + ((size_t)(nt * KT + kt) * 32 + lane) * 16;
  int k1w = K - ksplit;
  #pragma unroll
  for (int e = 0; e < 16; ++e){
    int k = kt * 32 + klocal(lane, e);
    float v = (k < ksplit) ? W0[(size_t)n * ksplit + k]
                           : W1[(size_t)n * k1w + (k - ksplit)];
    d[e] = f2bf(v);
  }
}

// ---------------------------- persistent decoder ---------------------------

__global__ __launch_bounds__(NTHR, 1) void k_decoder(DecParams P){
  const int tid  = threadIdx.x;
  const int gtid = blockIdx.x * NTHR + tid;
  const int wave = gtid >> 5;         // 0..255
  const int lane = tid & 31;
  unsigned epoch = 0;
  __shared__ float sE[TENC];
  __shared__ float sRed[2];
  __shared__ __align__(16) unsigned short sAct[B_ * LDA_D];   // 82KB, reused A<->D

  for (int t = 0; t < TDEC; ++t){
    // -------- stage actA -> LDS (async), overlapped with prev-step outputs ------
    for (int c = tid; c < B_ * (KA / 8); c += NTHR){
      int row = c / (KA / 8), cc = c - row * (KA / 8);
      async_g2l(sAct + (size_t)row * LDA_A + cc * 8, P.actA + (size_t)row * KA + cc * 8);
    }
    if (t > 0) do_outputs(P, gtid, t - 1);
    async_wait();
    __syncthreads();

    // -------- Phase A: attention-LSTM gates = [hpre|ctx|ah] @ [Wih;Whh]^T --------
    {
      v8f acc = gemm_tiles(sAct, LDA_A, P.wA, wave, KTA, lane);
      int n = wave * 16 + (lane & 15);
      store_rows(P.gates, NGATE, acc, wave, lane, P.arnn_bih[n] + P.arnn_bhh[n]);
    }
    grid_sync(P.cnt, epoch);

    // -------- Phase B: attention-LSTM pointwise --------
    for (int idx = gtid; idx < B_ * RNN; idx += NBLK * NTHR){
      int b = idx >> 10, u = idx & 1023;
      const float* g = P.gates + (size_t)b * NGATE;
      float gi = g[u], gf = g[RNN + u], gg = g[2 * RNN + u], go = g[3 * RNN + u];
      float c = sigm(gf) * P.ac[idx] + sigm(gi) * tanhf(gg);
      float h = sigm(go) * tanhf(c);
      P.ac[idx] = c;
      unsigned short hb = f2bf(h);
      P.actA[(size_t)b * KA + PRE + EMB + u] = hb;   // arnn recurrent input
      P.actD[(size_t)b * KD + u]             = hb;   // drnn feed-forward input
    }
    grid_sync(P.cnt, epoch);

    // -------- Phase C1: query projection (waves 0..7) || location conv+proj --------
    if (gtid < 256){                                  // block 0: pq = ah @ q_W.T + q_b
      v8f acc = gemm_tiles(P.actD, KD, P.wQ, wave, KTQ, lane);
      int n  = wave * 16 + (lane & 15);
      int mb = (lane < 16) ? 0 : 8;
      float qb = P.q_b[n];
      #pragma unroll
      for (int r = 0; r < 8; ++r) P.pq[(size_t)(mb + r) * ATT + n] = acc[r] + qb;
    } else if (gtid < 256 + B_ * TENC){               // 31-tap conv over [aw,awc] + ld proj
      int idx = gtid - 256;
      int b = idx >> 7, tp = idx & 127;
      float cc[NF];
      #pragma unroll
      for (int c = 0; c < NF; ++c) cc[c] = P.lc_b[c];
      for (int i = 0; i < 2; ++i){
        const float* src = (i == 0 ? P.aw : P.awc) + (size_t)b * TENC;
        for (int k = 0; k < KS; ++k){
          int p = tp - 15 + k;
          float xv = (p >= 0 && p < TENC) ? src[p] : 0.f;
          #pragma unroll
          for (int c = 0; c < NF; ++c) cc[c] += xv * P.lc_W[(c * 2 + i) * KS + k];
        }
      }
      float* dst = P.loc + ((size_t)b * TENC + tp) * ATT;
      for (int a = 0; a < ATT; ++a){
        float s = P.ld_b[a];
        const float* w = P.ld_W + (size_t)a * NF;
        #pragma unroll
        for (int c = 0; c < NF; ++c) s += cc[c] * w[c];
        dst[a] = s;
      }
    }
    grid_sync(P.cnt, epoch);

    // -------- Phase C2: energies + masked softmax + context (block b per batch) ----
    if (blockIdx.x < B_){
      int b = blockIdx.x;
      if (tid < TENC){
        int tp = tid;
        const float* loc = P.loc + ((size_t)b * TENC + tp) * ATT;
        const float* pm  = P.pm  + ((size_t)b * TENC + tp) * ATT;
        const float* pq  = P.pq  + (size_t)b * ATT;
        float s = 0.f;
        for (int a = 0; a < ATT; ++a) s += P.v_W[a] * tanhf(pq[a] + loc[a] + pm[a]);
        s += P.v_b[0];
        if (tp >= P.mlen[b]) s = -3.0e38f;
        sE[tp] = s;
      }
      __syncthreads();
      if (tid == 0){
        float mx = -3.0e38f;
        for (int i = 0; i < TENC; ++i) mx = fmaxf(mx, sE[i]);
        sRed[0] = mx;
      }
      __syncthreads();
      if (tid < TENC) sE[tid] = __expf(sE[tid] - sRed[0]);
      __syncthreads();
      if (tid == 0){
        float sm = 0.f;
        for (int i = 0; i < TENC; ++i) sm += sE[i];
        sRed[1] = sm;
      }
      __syncthreads();
      if (tid < TENC){
        float a = sE[tid] / sRed[1];
        sE[tid] = a;
        P.aw[(size_t)b * TENC + tid] = a;
        P.awc[(size_t)b * TENC + tid] += a;
        P.out_align[((size_t)b * TDEC + t) * TENC + tid] = a;
      }
      __syncthreads();
      for (int e = tid; e < EMB; e += NTHR){          // ctx = aw @ memory[b]
        const float* mem = P.memory + (size_t)b * TENC * EMB + e;
        float s = 0.f;
        for (int tp = 0; tp < TENC; ++tp) s += sE[tp] * mem[(size_t)tp * EMB];
        P.ctx[(size_t)b * EMB + e] = s;
        unsigned short cb = f2bf(s);
        P.actA[(size_t)b * KA + PRE + e] = cb;
        P.actD[(size_t)b * KD + RNN + e] = cb;
      }
    }
    grid_sync(P.cnt, epoch);

    // -------- stage actD -> LDS (async) --------
    for (int c = tid; c < B_ * (KD / 8); c += NTHR){
      int row = c / (KD / 8), cc = c - row * (KD / 8);
      async_g2l(sAct + (size_t)row * LDA_D + cc * 8, P.actD + (size_t)row * KD + cc * 8);
    }
    async_wait();
    __syncthreads();

    // -------- Phase D: decoder-LSTM gates = [ah|ctx|dh] @ [Wih;Whh]^T --------
    {
      v8f acc = gemm_tiles(sAct, LDA_D, P.wD, wave, KTD, lane);
      int n = wave * 16 + (lane & 15);
      store_rows(P.gates, NGATE, acc, wave, lane, P.drnn_bih[n] + P.drnn_bhh[n]);
    }
    grid_sync(P.cnt, epoch);

    // -------- Phase E: decoder-LSTM pointwise + stage next prenet frame --------
    for (int idx = gtid; idx < B_ * RNN; idx += NBLK * NTHR){
      int b = idx >> 10, u = idx & 1023;
      const float* g = P.gates + (size_t)b * NGATE;
      float gi = g[u], gf = g[RNN + u], gg = g[2 * RNN + u], go = g[3 * RNN + u];
      float c = sigm(gf) * P.dc[idx] + sigm(gi) * tanhf(gg);
      float h = sigm(go) * tanhf(c);
      P.dc[idx] = c;
      P.dh[idx] = h;
      P.actD[(size_t)b * KD + RNN + EMB + u] = f2bf(h);   // drnn recurrent input
    }
    if (t + 1 < TDEC && gtid < B_ * PRE){
      int b = gtid >> 8, j = gtid & 255;
      P.actA[(size_t)b * KA + j] = P.hpre[((size_t)(t + 1) * B_ + b) * PRE + j];
    }
    grid_sync(P.cnt, epoch);
  }
  do_outputs(P, gtid, TDEC - 1);
}

// ------------------------------- host side ---------------------------------

extern "C" void kernel_launch(void* const* d_in, const int* in_sizes, int n_in,
                              void* d_out, int out_size, void* d_ws, size_t ws_size,
                              hipStream_t stream){
  const float* memory   = (const float*)d_in[0];
  const float* din      = (const float*)d_in[1];
  const int*   mlen     = (const int*)  d_in[2];
  const float* pre_W1   = (const float*)d_in[3];
  const float* pre_b1   = (const float*)d_in[4];
  const float* pre_W2   = (const float*)d_in[5];
  const float* pre_b2   = (const float*)d_in[6];
  const float* arnn_Wih = (const float*)d_in[7];
  const float* arnn_Whh = (const float*)d_in[8];
  const float* arnn_bih = (const float*)d_in[9];
  const float* arnn_bhh = (const float*)d_in[10];
  const float* q_W      = (const float*)d_in[11];
  const float* q_b      = (const float*)d_in[12];
  const float* m_W      = (const float*)d_in[13];
  const float* m_b      = (const float*)d_in[14];
  const float* v_W      = (const float*)d_in[15];
  const float* v_b      = (const float*)d_in[16];
  const float* lc_W     = (const float*)d_in[17];
  const float* lc_b     = (const float*)d_in[18];
  const float* ld_W     = (const float*)d_in[19];
  const float* ld_b     = (const float*)d_in[20];
  const float* drnn_Wih = (const float*)d_in[21];
  const float* drnn_Whh = (const float*)d_in[22];
  const float* drnn_bih = (const float*)d_in[23];
  const float* drnn_bhh = (const float*)d_in[24];
  const float* proj_W   = (const float*)d_in[25];
  const float* proj_b   = (const float*)d_in[26];
  const float* gate_W   = (const float*)d_in[27];
  const float* gate_b   = (const float*)d_in[28];

  char* ws = (char*)d_ws;
  size_t off = 0;
  auto take = [&](size_t nbytes) -> char* {
    char* p = ws + off;
    off = (off + nbytes + 255) & ~(size_t)255;
    return p;
  };

  unsigned*       cnt  = (unsigned*)      take(256);
  unsigned short* wA   = (unsigned short*)take((size_t)NGATE * KA * 2);
  unsigned short* wD   = (unsigned short*)take((size_t)NGATE * KD * 2);
  unsigned short* wQ   = (unsigned short*)take((size_t)ATT * RNN * 2);
  unsigned short* hpre = (unsigned short*)take((size_t)TDEC * B_ * PRE * 2);
  float*          l1   = (float*)         take((size_t)TDEC * B_ * PRE * 4);
  float*          pm   = (float*)         take((size_t)B_ * TENC * ATT * 4);
  float*          gts  = (float*)         take((size_t)B_ * NGATE * 4);
  unsigned short* actA = (unsigned short*)take((size_t)B_ * KA * 2);
  unsigned short* actD = (unsigned short*)take((size_t)B_ * KD * 2);
  float*          ac   = (float*)         take((size_t)B_ * RNN * 4);
  float*          dc   = (float*)         take((size_t)B_ * RNN * 4);
  float*          dh   = (float*)         take((size_t)B_ * RNN * 4);
  float*          aw   = (float*)         take((size_t)B_ * TENC * 4);
  float*          awc  = (float*)         take((size_t)B_ * TENC * 4);
  float*          ctx  = (float*)         take((size_t)B_ * EMB * 4);
  float*          loc  = (float*)         take((size_t)B_ * TENC * ATT * 4);
  float*          pq   = (float*)         take((size_t)B_ * ATT * 4);
  (void)ws_size; (void)in_sizes; (void)n_in; (void)out_size;

  DecParams P;
  P.memory = memory; P.din = din; P.mlen = mlen;
  P.arnn_bih = arnn_bih; P.arnn_bhh = arnn_bhh;
  P.drnn_bih = drnn_bih; P.drnn_bhh = drnn_bhh;
  P.q_b = q_b; P.v_W = v_W; P.v_b = v_b;
  P.lc_W = lc_W; P.lc_b = lc_b; P.ld_W = ld_W; P.ld_b = ld_b;
  P.proj_W = proj_W; P.proj_b = proj_b; P.gate_W = gate_W; P.gate_b = gate_b;
  P.wA = wA; P.wD = wD; P.wQ = wQ; P.hpre = hpre;
  P.actA = actA; P.actD = actD;
  P.gates = gts; P.ac = ac; P.dc = dc; P.dh = dh;
  P.aw = aw; P.awc = awc; P.ctx = ctx; P.loc = loc; P.pq = pq; P.pm = pm;
  P.cnt = cnt;
  P.out_mel   = (float*)d_out;
  P.out_gate  = (float*)d_out + (size_t)B_ * MEL * TDEC;
  P.out_align = (float*)d_out + (size_t)B_ * MEL * TDEC + (size_t)B_ * TDEC;

  // state init (also zeroes the barrier counter -> deterministic graph replay)
  k_zero<<<128, 256, 0, stream>>>(P);
  // prenet (hoisted out of the scan)
  k_prenet1<<<(TDEC * B_ * PRE + 255) / 256, 256, 0, stream>>>(din, pre_W1, pre_b1, l1);
  k_prenet2<<<(TDEC * B_ * PRE + 255) / 256, 256, 0, stream>>>(l1, pre_W2, pre_b2, hpre, actA);
  // memory projection (hoisted)
  k_pm<<<(B_ * TENC * ATT + 255) / 256, 256, 0, stream>>>(memory, m_W, m_b, pm);
  // weight conversion + WMMA swizzle (weights then persist in L2 for all 200 steps)
  k_swizzle<<<((NGATE / 16) * KTA * 32 + 255) / 256, 256, 0, stream>>>(arnn_Wih, arnn_Whh, PRE + EMB, KA, NGATE, wA);
  k_swizzle<<<((NGATE / 16) * KTD * 32 + 255) / 256, 256, 0, stream>>>(drnn_Wih, drnn_Whh, RNN + EMB, KD, NGATE, wD);
  k_swizzle<<<((ATT / 16) * KTQ * 32 + 255) / 256, 256, 0, stream>>>(q_W, q_W, RNN, RNN, ATT, wQ);
  // persistent 200-step decoder scan
  k_decoder<<<NBLK, NTHR, 0, stream>>>(P);
}
